// DGCNNClassification_28965259444561
// MI455X (gfx1250) — matrix operand order repaired and checked by reading, not compiled, not run on hardware
//
#include <hip/hip_runtime.h>

typedef _Float16 half_t;
typedef __attribute__((ext_vector_type(16))) _Float16 v16h;
typedef __attribute__((ext_vector_type(8)))  _Float16 v8h;
typedef __attribute__((ext_vector_type(8)))  float    v8f;

#define B_    32
#define N_    1024
#define K_    20
#define NPTS  32768   // B_ * N_

// ---------------------------------------------------------------------------
// WMMA helpers (CDNA5 gfx1250, wave32): D = A(16x32 f16) * B(32x16 f16) + C(f32)
// ---------------------------------------------------------------------------
__device__ __forceinline__ v8f wmma16(v16h a, v16h b, v8f c) {
  return __builtin_amdgcn_wmma_f32_16x16x32_f16(
      /*neg_a=*/false, a, /*neg_b=*/false, b,
      /*c_mod=*/(short)0, c, /*reuse_a=*/false, /*reuse_b=*/false);
}

__device__ __forceinline__ v16h cat8(v8h lo, v8h hi) {
  return __builtin_shufflevector(lo, hi, 0, 1, 2, 3, 4, 5, 6, 7,
                                         8, 9, 10, 11, 12, 13, 14, 15);
}

// A fragment from a row-major 16xKD tile (LDS or global), 16B-aligned rows.
// Lane L holds row L%16; chunks: lanes<16 -> K=[0..7],[16..23]; lanes>=16 -> [8..15],[24..31].
__device__ __forceinline__ v16h load_a_tile(const half_t* tile, int ld, int lane, int kbase) {
  const int row = lane & 15;
  const int b0  = (lane < 16) ? 0 : 8;
  const v8h* p = (const v8h*)(tile + row * ld + kbase + b0);   // 16B aligned
  v8h lo = p[0];      // K = base .. base+7
  v8h hi = p[2];      // K = base+16 .. base+23
  return cat8(lo, hi);
}

// B fragment from pre-swizzled (packed) weights: lane's 16 halves are contiguous.
// Packed layout: [blk][lane][e], blk = kblk * (ncol/16) + ntile.
__device__ __forceinline__ v16h load_b_packed(const half_t* Wpk, int blk, int lane) {
  const v8h* p = (const v8h*)(Wpk + ((size_t)blk * 32 + lane) * 16);  // 32B/lane
  return cat8(p[0], p[1]);
}

// Float max via integer atomic — valid for non-negative values, 0-initialized dest.
__device__ __forceinline__ void atomic_max_f32(float* p, float v) {
  atomicMax((int*)p, __float_as_int(v));
}

// ---------------------------------------------------------------------------
// Utility kernels
// ---------------------------------------------------------------------------
__global__ void dg_zero_f32(float* __restrict__ p, int n) {
  int g = blockIdx.x * blockDim.x + threadIdx.x;
  if (g < n) p[g] = 0.f;
}

__global__ void dg_cvt_f16(const float* __restrict__ s, half_t* __restrict__ d, int n) {
  int g = blockIdx.x * blockDim.x + threadIdx.x;
  if (g < n) d[g] = (half_t)s[g];
}

// f32 [kin x ncol] -> f16 packed into B-fragment layout, K zero-padded to kpad.
// Element e of lane (hi*16 + col%16) in block (k/32)*(ncol/16) + col/16 holds
// W[(k) * ncol + col] with k = 32*kblk + (e<8 ? 8*hi + e : 16 + 8*hi + e-8).
__global__ void dg_pack_b(const float* __restrict__ src, half_t* __restrict__ dst,
                          int kin, int kpad, int ncol) {
  int g = blockIdx.x * blockDim.x + threadIdx.x;
  if (g >= kpad * ncol) return;
  int k = g / ncol, col = g - k * ncol;
  int kblk = k >> 5, krem = k & 31;
  int hi = (krem >> 3) & 1, e = (krem & 7) | ((krem >> 4) << 3);
  int ntile = col >> 4, lc = col & 15;
  int lane = hi * 16 + lc;
  int ntiles = ncol >> 4;
  size_t di = (((size_t)(kblk * ntiles + ntile) * 32) + lane) * 16 + e;
  dst[di] = (k < kin) ? (half_t)src[k * ncol + col] : (half_t)0.f;
}

// ---------------------------------------------------------------------------
// kNN (brute force, top-20 insertion sort per point)
// ---------------------------------------------------------------------------
__global__ __launch_bounds__(256) void dg_knn3(const float* __restrict__ x, int* __restrict__ idx) {
  int g = blockIdx.x * blockDim.x + threadIdx.x;
  if (g >= NPTS) return;
  int n = g & (N_ - 1), base = g & ~(N_ - 1);
  float xi0 = x[g * 3 + 0], xi1 = x[g * 3 + 1], xi2 = x[g * 3 + 2];
  float bd[K_]; int bi[K_];
#pragma unroll
  for (int s = 0; s < K_; ++s) { bd[s] = 3.4e38f; bi[s] = base; }
  for (int j = 0; j < N_; ++j) {
    if (j == n) continue;
    int gj = base + j;
    float d0 = x[gj * 3 + 0] - xi0, d1 = x[gj * 3 + 1] - xi1, d2 = x[gj * 3 + 2] - xi2;
    float d = d0 * d0 + d1 * d1 + d2 * d2;
    if (d < bd[K_ - 1]) {
      int s = K_ - 1;
      while (s > 0 && bd[s - 1] > d) { bd[s] = bd[s - 1]; bi[s] = bi[s - 1]; --s; }
      bd[s] = d; bi[s] = gj;
    }
  }
  for (int s = 0; s < K_; ++s) idx[g * K_ + s] = bi[s];
}

__global__ __launch_bounds__(256) void dg_knn64(const float* __restrict__ f, int* __restrict__ idx) {
  int g = blockIdx.x * blockDim.x + threadIdx.x;
  if (g >= NPTS) return;
  int n = g & (N_ - 1), base = g & ~(N_ - 1);
  float xi[64];
#pragma unroll
  for (int c = 0; c < 64; ++c) xi[c] = f[g * 64 + c];
  float bd[K_]; int bi[K_];
#pragma unroll
  for (int s = 0; s < K_; ++s) { bd[s] = 3.4e38f; bi[s] = base; }
  for (int j = 0; j < N_; ++j) {
    if (j == n) continue;
    const float* pj = f + (size_t)(base + j) * 64;
    float d = 0.f;
#pragma unroll 16
    for (int c = 0; c < 64; ++c) { float t = pj[c] - xi[c]; d += t * t; }
    if (d < bd[K_ - 1]) {
      int s = K_ - 1;
      while (s > 0 && bd[s - 1] > d) { bd[s] = bd[s - 1]; bi[s] = bi[s - 1]; --s; }
      bd[s] = d; bi[s] = base + j;
    }
  }
  for (int s = 0; s < K_; ++s) idx[g * K_ + s] = bi[s];
}

// ---------------------------------------------------------------------------
// EdgeConv1: fused gather + 3-layer MLP (6->64->64->64, ReLU) + max-over-k.
// Block = 16 points = 320 edges = 20 M-tiles of 16; 8 waves stride over tiles.
// Weights arrive pre-packed in B-fragment layout.
// ---------------------------------------------------------------------------
__global__ __launch_bounds__(256) void dg_edgeconv1(
    const float* __restrict__ x, const int* __restrict__ idx,
    const half_t* __restrict__ W1p, const float* __restrict__ b1,
    const half_t* __restrict__ W2p, const float* __restrict__ b2,
    const half_t* __restrict__ W3p, const float* __restrict__ b3,
    float* __restrict__ out1)
{
  __shared__ __align__(16) half_t sW1[32 * 64];    // 1 kblk * 4 ntiles
  __shared__ __align__(16) half_t sW2[64 * 64];    // 2 kblks * 4 ntiles
  __shared__ __align__(16) half_t sW3[64 * 64];
  __shared__ float  sB[3 * 64];
  __shared__ __align__(16) half_t sHa[8][16 * 64]; // per-wave inter-layer tiles
  __shared__ __align__(16) half_t sHb[8][16 * 64];

  const int tid = threadIdx.x, wave = tid >> 5, lane = tid & 31;
  const int pbase = blockIdx.x * 16;

  for (int t = tid; t < (32 * 64) / 8; t += 256) ((v8h*)sW1)[t] = ((const v8h*)W1p)[t];
  for (int t = tid; t < (64 * 64) / 8; t += 256) ((v8h*)sW2)[t] = ((const v8h*)W2p)[t];
  for (int t = tid; t < (64 * 64) / 8; t += 256) ((v8h*)sW3)[t] = ((const v8h*)W3p)[t];
  for (int t = tid; t < 64; t += 256) { sB[t] = b1[t]; sB[64 + t] = b2[t]; sB[128 + t] = b3[t]; }
  __syncthreads();

  const int colL = lane & 15;
  const int rb   = (lane < 16) ? 0 : 8;

  for (int t = wave; t < 20; t += 8) {
    // ---- layer 1: A = [x_i, x_j - x_i] (K padded 6->32), fused gather ----
    v16h a1;
#pragma unroll
    for (int e = 0; e < 16; ++e) a1[e] = (half_t)0.f;
    if (lane < 16) {   // lanes>=16 own K=8..15,24..31 which are all padding
      int r    = t * 16 + lane;
      int pl   = r / K_;
      int slot = r - pl * K_;
      int i    = pbase + pl;
      int j    = idx[i * K_ + slot];
      float xi0 = x[i * 3 + 0], xi1 = x[i * 3 + 1], xi2 = x[i * 3 + 2];
      float xj0 = x[j * 3 + 0], xj1 = x[j * 3 + 1], xj2 = x[j * 3 + 2];
      a1[0] = (half_t)xi0; a1[1] = (half_t)xi1; a1[2] = (half_t)xi2;
      a1[3] = (half_t)(xj0 - xi0); a1[4] = (half_t)(xj1 - xi1); a1[5] = (half_t)(xj2 - xi2);
    }
#pragma unroll
    for (int nt = 0; nt < 4; ++nt) {
      v8f acc = {};
      acc = wmma16(a1, load_b_packed(sW1, nt, lane), acc);
      float bias = sB[nt * 16 + colL];
#pragma unroll
      for (int v = 0; v < 8; ++v) {
        float h = fmaxf(acc[v] + bias, 0.f);
        sHa[wave][(rb + v) * 64 + nt * 16 + colL] = (half_t)h;
      }
    }
    // ---- layer 2: 64 -> 64 ----
#pragma unroll
    for (int nt = 0; nt < 4; ++nt) {
      v8f acc = {};
#pragma unroll
      for (int kb = 0; kb < 2; ++kb) {
        v16h a = load_a_tile(&sHa[wave][0], 64, lane, kb * 32);
        acc = wmma16(a, load_b_packed(sW2, kb * 4 + nt, lane), acc);
      }
      float bias = sB[64 + nt * 16 + colL];
#pragma unroll
      for (int v = 0; v < 8; ++v) {
        float h = fmaxf(acc[v] + bias, 0.f);
        sHb[wave][(rb + v) * 64 + nt * 16 + colL] = (half_t)h;
      }
    }
    // ---- layer 3: 64 -> 64, ReLU, max-over-k via f32-bit atomics ----
#pragma unroll
    for (int nt = 0; nt < 4; ++nt) {
      v8f acc = {};
#pragma unroll
      for (int kb = 0; kb < 2; ++kb) {
        v16h a = load_a_tile(&sHb[wave][0], 64, lane, kb * 32);
        acc = wmma16(a, load_b_packed(sW3, kb * 4 + nt, lane), acc);
      }
      float bias = sB[128 + nt * 16 + colL];
#pragma unroll
      for (int v = 0; v < 8; ++v) {
        float h = fmaxf(acc[v] + bias, 0.f);
        int r = t * 16 + rb + v;
        int p = pbase + r / K_;
        atomic_max_f32(&out1[(size_t)p * 64 + nt * 16 + colL], h);
      }
    }
  }
}

// ---------------------------------------------------------------------------
// EdgeConv2: fused gather + (128->128, ReLU) + max-over-k.
// ---------------------------------------------------------------------------
__global__ __launch_bounds__(256) void dg_edgeconv2(
    const half_t* __restrict__ F, const int* __restrict__ idx,
    const half_t* __restrict__ W4p, const float* __restrict__ b4,
    float* __restrict__ out2)
{
  __shared__ __align__(16) half_t sW4[128 * 128];  // 4 kblks * 8 ntiles
  __shared__ float  sB[128];
  const int tid = threadIdx.x, wave = tid >> 5, lane = tid & 31;
  const int pbase = blockIdx.x * 16;

  for (int t = tid; t < (128 * 128) / 8; t += 256) ((v8h*)sW4)[t] = ((const v8h*)W4p)[t];
  for (int t = tid; t < 128; t += 256) sB[t] = b4[t];
  __syncthreads();

  const int colL = lane & 15;
  const int rb   = (lane < 16) ? 0 : 8;

  for (int t = wave; t < 20; t += 8) {
    const int r    = t * 16 + (lane & 15);
    const int pl   = r / K_;
    const int i    = pbase + pl;
    const int j    = idx[i * K_ + (r - pl * K_)];
    const half_t* Fi = F + (size_t)i * 64;
    const half_t* Fj = F + (size_t)j * 64;

    // 4 A fragments covering K=128: A(:,k) = k<64 ? f_i[k] : f_j[k-64]-f_i[k-64]
    const int b0 = (lane < 16) ? 0 : 8;
    v16h afr[4];
#pragma unroll
    for (int c = 0; c < 4; ++c) {
      v8h part[2];
#pragma unroll
      for (int hs = 0; hs < 2; ++hs) {
        int base = c * 32 + b0 + hs * 16;   // 8-aligned, never straddles k=64
        v8h v;
        if (base < 64) {
          v = *(const v8h*)(Fi + base);
        } else {
          v = *(const v8h*)(Fj + (base - 64)) - *(const v8h*)(Fi + (base - 64));
        }
        part[hs] = v;
      }
      afr[c] = cat8(part[0], part[1]);
    }
#pragma unroll
    for (int nt = 0; nt < 8; ++nt) {
      v8f acc = {};
#pragma unroll
      for (int c = 0; c < 4; ++c)
        acc = wmma16(afr[c], load_b_packed(sW4, c * 8 + nt, lane), acc);
      float bias = sB[nt * 16 + colL];
#pragma unroll
      for (int v = 0; v < 8; ++v) {
        float h = fmaxf(acc[v] + bias, 0.f);
        int rr = t * 16 + rb + v;
        int p  = pbase + rr / K_;
        atomic_max_f32(&out2[(size_t)p * 128 + nt * 16 + colL], h);
      }
    }
  }
}

// ---------------------------------------------------------------------------
// relu(X[32768,128] @ Wp[128,512] + bp) fused with global max-pool over N.
// grid = (256, 32): blockIdx.x -> 8 M-tiles (one per wave), blockIdx.y -> N-tile.
// ---------------------------------------------------------------------------
__global__ __launch_bounds__(256) void dg_gemm_pool(
    const half_t* __restrict__ Xp, const half_t* __restrict__ Wpk,
    const float* __restrict__ bp, float* __restrict__ pool)
{
  __shared__ __align__(16) half_t sW[4 * 32 * 16]; // 4 kblks, 1 ntile slice
  const int tid = threadIdx.x, wave = tid >> 5, lane = tid & 31;
  const int ntile = blockIdx.y;                    // 0..31

  // Stage this block's 16-column slice of packed Wp: blk = kblk*32 + ntile.
  for (int t = tid; t < 4 * 64; t += 256) {        // v8h granularity
    int kblk = t >> 6, r = t & 63;
    ((v8h*)sW)[t] = ((const v8h*)Wpk)[(size_t)(kblk * 32 + ntile) * 64 + r];
  }
  __syncthreads();

  const int colL = lane & 15, rb = (lane < 16) ? 0 : 8;
  const int mtile = blockIdx.x * 8 + wave;         // 2048 M-tiles
  const half_t* Atile = Xp + (size_t)mtile * 16 * 128;
  __builtin_prefetch(Atile, 0, 0);                 // global_prefetch_b8

  v8f acc = {};
#pragma unroll
  for (int kb = 0; kb < 4; ++kb) {
    v16h a = load_a_tile(Atile, 128, lane, kb * 32);
    acc = wmma16(a, load_b_packed(sW, kb, lane), acc);
  }
  float bias = bp[ntile * 16 + colL];
#pragma unroll
  for (int v = 0; v < 8; ++v) {
    float h = fmaxf(acc[v] + bias, 0.f);
    int gr = mtile * 16 + rb + v;
    int bi = gr >> 10;                             // batch index
    atomic_max_f32(&pool[(size_t)bi * 512 + ntile * 16 + colL], h);
  }
}

// ---------------------------------------------------------------------------
// Head: out[b,:] = relu(pool[b,:] @ Wt1 + bt1) @ Wt2 + bt2   (M=32, scalar)
// ---------------------------------------------------------------------------
__global__ __launch_bounds__(256) void dg_head(
    const float* __restrict__ pool,
    const float* __restrict__ Wt1, const float* __restrict__ bt1,
    const float* __restrict__ Wt2, const float* __restrict__ bt2,
    float* __restrict__ out)
{
  __shared__ float sIn[512];
  __shared__ float sT[256];
  const int b = blockIdx.x, tid = threadIdx.x;
  for (int t = tid; t < 512; t += 256) sIn[t] = pool[b * 512 + t];
  __syncthreads();
  float acc = bt1[tid];
  for (int k = 0; k < 512; ++k) acc += sIn[k] * Wt1[k * 256 + tid];
  sT[tid] = fmaxf(acc, 0.f);
  __syncthreads();
  if (tid < 40) {
    float o = bt2[tid];
    for (int k = 0; k < 256; ++k) o += sT[k] * Wt2[k * 40 + tid];
    out[b * 40 + tid] = o;
  }
}

// ---------------------------------------------------------------------------
// Host launcher
// ---------------------------------------------------------------------------
extern "C" void kernel_launch(void* const* d_in, const int* in_sizes, int n_in,
                              void* d_out, int out_size, void* d_ws, size_t ws_size,
                              hipStream_t stream) {
  (void)in_sizes; (void)n_in; (void)out_size; (void)ws_size;

  const float* x   = (const float*)d_in[0];
  // d_in[1] = batch (unused: equal-sized clouds)
  const float* W1  = (const float*)d_in[2];
  const float* b1  = (const float*)d_in[3];
  const float* W2  = (const float*)d_in[4];
  const float* b2  = (const float*)d_in[5];
  const float* W3  = (const float*)d_in[6];
  const float* b3  = (const float*)d_in[7];
  const float* W4  = (const float*)d_in[8];
  const float* b4  = (const float*)d_in[9];
  const float* Wp  = (const float*)d_in[10];
  const float* bp  = (const float*)d_in[11];
  const float* Wt1 = (const float*)d_in[12];
  const float* bt1 = (const float*)d_in[13];
  const float* Wt2 = (const float*)d_in[14];
  const float* bt2 = (const float*)d_in[15];
  float* out = (float*)d_out;

  char* ws = (char*)d_ws;
  size_t off = 0;
  auto alloc = [&](size_t bytes) -> void* {
    void* p = (void*)(ws + off);
    off += (bytes + 255) & ~(size_t)255;
    return p;
  };
  int*    idx1 = (int*)   alloc((size_t)NPTS * K_ * 4);
  int*    idx2 = (int*)   alloc((size_t)NPTS * K_ * 4);
  float*  out1 = (float*) alloc((size_t)NPTS * 64 * 4);
  float*  out2 = (float*) alloc((size_t)NPTS * 128 * 4);
  half_t* F1h  = (half_t*)alloc((size_t)NPTS * 64 * 2);
  half_t* F2h  = (half_t*)alloc((size_t)NPTS * 128 * 2);
  float*  pool = (float*) alloc((size_t)B_ * 512 * 4);
  half_t* W1p  = (half_t*)alloc(32 * 64 * 2);
  half_t* W2p  = (half_t*)alloc(64 * 64 * 2);
  half_t* W3p  = (half_t*)alloc(64 * 64 * 2);
  half_t* W4p  = (half_t*)alloc(128 * 128 * 2);
  half_t* Wpp  = (half_t*)alloc(128 * 512 * 2);

  // Zero the atomic-max accumulators.
  dg_zero_f32<<<(NPTS * 64 + 255) / 256, 256, 0, stream>>>(out1, NPTS * 64);
  dg_zero_f32<<<(NPTS * 128 + 255) / 256, 256, 0, stream>>>(out2, NPTS * 128);
  dg_zero_f32<<<(B_ * 512 + 255) / 256, 256, 0, stream>>>(pool, B_ * 512);

  // Weight conversion: f32 -> f16, swizzled into B-fragment layout, K-padded.
  dg_pack_b<<<(32 * 64 + 255) / 256, 256, 0, stream>>>(W1, W1p, 6, 32, 64);
  dg_pack_b<<<(64 * 64 + 255) / 256, 256, 0, stream>>>(W2, W2p, 64, 64, 64);
  dg_pack_b<<<(64 * 64 + 255) / 256, 256, 0, stream>>>(W3, W3p, 64, 64, 64);
  dg_pack_b<<<(128 * 128 + 255) / 256, 256, 0, stream>>>(W4, W4p, 128, 128, 128);
  dg_pack_b<<<(128 * 512 + 255) / 256, 256, 0, stream>>>(Wp, Wpp, 128, 128, 512);

  // Stage 1
  dg_knn3<<<NPTS / 256, 256, 0, stream>>>(x, idx1);
  dg_edgeconv1<<<NPTS / 16, 256, 0, stream>>>(x, idx1, W1p, b1, W2p, b2, W3p, b3, out1);
  dg_cvt_f16<<<(NPTS * 64 + 255) / 256, 256, 0, stream>>>(out1, F1h, NPTS * 64);

  // Stage 2
  dg_knn64<<<NPTS / 256, 256, 0, stream>>>(out1, idx2);
  dg_edgeconv2<<<NPTS / 16, 256, 0, stream>>>(F1h, idx2, W4p, b4, out2);
  dg_cvt_f16<<<(NPTS * 128 + 255) / 256, 256, 0, stream>>>(out2, F2h, NPTS * 128);

  // Projection + global max pool
  dg_gemm_pool<<<dim3(NPTS / 16 / 8, 512 / 16), 256, 0, stream>>>(F2h, Wpp, bp, pool);

  // Classifier head
  dg_head<<<B_, 256, 0, stream>>>(pool, Wt1, bt1, Wt2, bt2, out);
}
